// CausalSelfAttention_16810501996824
// MI455X (gfx1250) — compile-verified, matrix operand order checked by reading, not run on hardware
//
#include <hip/hip_runtime.h>

typedef __attribute__((ext_vector_type(16))) _Float16 v16h;
typedef __attribute__((ext_vector_type(8)))  float    v8f;
typedef __attribute__((ext_vector_type(4)))  unsigned u32x4;
typedef __attribute__((ext_vector_type(8)))  unsigned u32x8;

#define DEVINL static __device__ __forceinline__

namespace {

constexpr int T  = 2048;   // sequence length
constexpr int D  = 1024;   // model dim
constexpr int H  = 16;     // heads
constexpr int HD = 64;     // head dim

DEVINL v8f wmma16(v16h a, v16h b, v8f c) {
  // D = A(16x32 f16) * B(32x16 f16) + C(16x16 f32)
  return __builtin_amdgcn_wmma_f32_16x16x32_f16(false, a, false, b, (short)0, c,
                                                false, false);
}

DEVINL v8f vzero8() {
  v8f z;
#pragma unroll
  for (int i = 0; i < 8; ++i) z[i] = 0.f;
  return z;
}

// Load a 16x32 f16 tile (row-major, leading dim `ld`) into the WMMA A-fragment
// layout (ISA 7.12.2): lane L<16 holds row L, K in {0..7}u{16..23};
// lane L>=16 holds row L-16, K in {8..15}u{24..31}; 2 halfs per VGPR.
// Passing such a fragment as the B operand multiplies by the tile's TRANSPOSE.
DEVINL v16h frag_a(const _Float16* tile, int ld) {
  const int lane = threadIdx.x & 31;
  const int m    = lane & 15;
  const int kb   = (lane & 16) >> 1;  // 0 or 8
  const _Float16* row = tile + m * ld + kb;
  union { v16h v; unsigned u[8]; } r;
#pragma unroll
  for (int i = 0; i < 4; ++i) {
    r.u[i]     = *reinterpret_cast<const unsigned*>(row + 2 * i);
    r.u[i + 4] = *reinterpret_cast<const unsigned*>(row + 16 + 2 * i);
  }
  return r.v;
}

// ---- Tensor Data Mover: 2D f16 tile (dim0 elems x dim1 rows) global -> LDS.
// D# packing per CDNA5 ISA. Issue once per wave.
DEVINL void tdm_load_2d_f16(unsigned lds_addr, const void* gsrc,
                            unsigned dim0, unsigned dim1, unsigned stride0) {
  const unsigned long long ga = (unsigned long long)(uintptr_t)gsrc;
  u32x4 g0;
  g0[0] = 1u;                                   // count=1, user D#, no gather
  g0[1] = lds_addr;                             // lds byte address
  g0[2] = (unsigned)(ga & 0xffffffffu);         // global_addr[31:0]
  g0[3] = (unsigned)((ga >> 32) & 0x01ffffffu)  // global_addr[56:32]
          | (2u << 30);                         // type=2 (image)
  u32x8 g1;
  g1[0] = 1u << 16;                             // data_size=1 (2B); mask=0
  g1[1] = (dim0 & 0xffffu) << 16;               // tensor_dim0[15:0]
  g1[2] = ((dim0 >> 16) & 0xffffu) | ((dim1 & 0xffffu) << 16);  // dim0 hi, dim1 lo
  g1[3] = ((dim1 >> 16) & 0xffffu) | ((dim0 & 0xffffu) << 16);  // dim1 hi, tile_dim0
  g1[4] = dim1 & 0xffffu;                       // tile_dim1, tile_dim2=0
  g1[5] = stride0;                              // tensor_dim0_stride[31:0]
  g1[6] = 0u;                                   // stride0 hi, stride1 lo
  g1[7] = 0u;
  asm volatile("tensor_load_to_lds %0, %1" :: "s"(g0), "s"(g1) : "memory");
}

// ---- Async global -> LDS staging of `bytes` (multiple of blockDim*16),
// cooperative across the block, 16B per lane-op. Tracked by ASYNCcnt.
DEVINL void async_stage(const void* gsrc, void* lds, int bytes) {
  const unsigned lbase = (unsigned)(uintptr_t)lds;
  const unsigned long long gbase = (unsigned long long)(uintptr_t)gsrc;
  for (unsigned off = threadIdx.x * 16u; off < (unsigned)bytes;
       off += blockDim.x * 16u) {
    const unsigned la = lbase + off;
    asm volatile("global_load_async_to_lds_b128 %0, %1, %2"
                 :: "v"(la), "v"(off), "s"(gbase) : "memory");
  }
  asm volatile("s_wait_asynccnt 0x0" ::: "memory");
}

// C/D fragment layout (f32 16x16): lane l holds col = l&15; VGPR r holds
// row = r + 8*(l>=16).

// RMS-norm over head dim (64) + RoPE + f16 store of a 16x64 tile held in 4
// C-fragments. dst_head points at [t][64] rows of this head.
DEVINL void rmsnorm_rope_store(const v8f* acc, _Float16* dst_head, int t0) {
  const int lane = threadIdx.x & 31;
  const int col  = lane & 15;
  const int hi8  = (lane & 16) >> 1;
  float ss[8];
#pragma unroll
  for (int r = 0; r < 8; ++r) {
    float s = 0.f;
#pragma unroll
    for (int j = 0; j < 4; ++j) { float q = acc[j][r]; s += q * q; }
    ss[r] = s;
  }
#pragma unroll
  for (int msk = 1; msk < 16; msk <<= 1) {
#pragma unroll
    for (int r = 0; r < 8; ++r) ss[r] += __shfl_xor(ss[r], msk, 32);
  }
#pragma unroll
  for (int r = 0; r < 8; ++r) {
    const float rinv = rsqrtf(ss[r] * (1.f / 64.f) + 1e-6f);
    const int   t    = t0 + r + hi8;
#pragma unroll
    for (int j = 0; j < 2; ++j) {
      const int   i    = j * 16 + col;  // 0..31 (rotary half-index)
      const float invf = __powf(10000.f, -(float)(2 * i) * (1.f / 64.f));
      float sn, cs;
      __sincosf((float)t * invf, &sn, &cs);
      const float x1 = acc[j][r]     * rinv;
      const float x2 = acc[j + 2][r] * rinv;   // pair (i, i+32): same lane!
      dst_head[t * HD + i]      = (_Float16)( x1 * cs + x2 * sn);
      dst_head[t * HD + i + 32] = (_Float16)(-x1 * sn + x2 * cs);
    }
  }
}

__global__ void __launch_bounds__(256) convert_f16(const float* __restrict__ in,
                                                   _Float16* __restrict__ out,
                                                   int n) {
  int i = blockIdx.x * blockDim.x + threadIdx.x;
  if (i < n) out[i] = (_Float16)in[i];
}

// ---------------- Phase 1: fused QKV + v-blend + rmsnorm + rope + gate ------
// grid (T/16, 4), block 128 (4 waves). Wave w handles head h = by*4+w for a
// 16-row t-tile. The x tile (16x1024 f16, 32 KB) is shared by all 4 waves:
// staged once into LDS by the Tensor Data Mover, A-fragments then come from
// LDS while W B-fragments stream from global/L2.
__global__ void __launch_bounds__(128) qkv_kernel(
    const float* __restrict__ x, const float* __restrict__ v1,
    const float* __restrict__ lambp, const float* __restrict__ Wgate,
    const _Float16* __restrict__ xh, const _Float16* __restrict__ wq,
    const _Float16* __restrict__ wk, const _Float16* __restrict__ wv,
    _Float16* __restrict__ qh, _Float16* __restrict__ kh,
    _Float16* __restrict__ vT, float* __restrict__ gate) {
  __shared__ __align__(16) _Float16 xtile[16 * D];   // 32 KB
  const int w    = threadIdx.x >> 5;
  const int lane = threadIdx.x & 31;
  const int h    = blockIdx.y * 4 + w;
  const int t0   = blockIdx.x * 16;
  const int col  = lane & 15;
  const int hi8  = (lane & 16) >> 1;

  if (threadIdx.x < 32) {  // wave 0 issues one TDM DMA for the whole block
    tdm_load_2d_f16((unsigned)(uintptr_t)&xtile[0], xh + t0 * D,
                    /*dim0=*/D, /*dim1=*/16, /*stride0=*/D);
    __builtin_amdgcn_s_wait_tensorcnt(0);
  }
  __syncthreads();

  v8f aq[4], ak[4], av[4];
#pragma unroll
  for (int j = 0; j < 4; ++j) { aq[j] = vzero8(); ak[j] = vzero8(); av[j] = vzero8(); }

  for (int k0 = 0; k0 < D; k0 += 32) {
    const v16h a = frag_a(xtile + k0, D);          // ds_load from LDS
#pragma unroll
    for (int j = 0; j < 4; ++j) {
      const int n0 = h * HD + j * 16;
      v16h b;
      b = frag_a(wq + n0 * D + k0, D); aq[j] = wmma16(a, b, aq[j]);
      b = frag_a(wk + n0 * D + k0, D); ak[j] = wmma16(a, b, ak[j]);
      b = frag_a(wv + n0 * D + k0, D); av[j] = wmma16(a, b, av[j]);
    }
  }

  // v = (1-lamb)*v + lamb*v1, stored TRANSPOSED per head: vT[h][d][s]
  const float lamb = *lambp;
#pragma unroll
  for (int j = 0; j < 4; ++j) {
#pragma unroll
    for (int r = 0; r < 8; ++r) {
      const int t = t0 + r + hi8;
      const int d = j * 16 + col;
      const float vv = (1.f - lamb) * av[j][r] + lamb * v1[t * D + h * HD + d];
      vT[(h * HD + d) * T + t] = (_Float16)vv;
    }
  }

  rmsnorm_rope_store(aq, qh + h * T * HD, t0);
  rmsnorm_rope_store(ak, kh + h * T * HD, t0);

  // gate[t][h] = sigmoid(x[t,:12] . Wgate[h,:])
  if (lane < 16) {
    const int t = t0 + lane;
    float s = 0.f;
#pragma unroll
    for (int jj = 0; jj < 12; ++jj) s += x[t * D + jj] * Wgate[h * 12 + jj];
    gate[t * H + h] = 1.f / (1.f + __expf(-s));
  }
}

// ---------------- Phase 2: causal flash attention ---------------------------
// grid (T/16, 4), block 128. Wave handles one (16-row q-tile, head); streams
// s in blocks of 32 with online softmax; P goes C-layout -> A-layout via LDS.
__global__ void __launch_bounds__(128) attn_kernel(
    const _Float16* __restrict__ qh, const _Float16* __restrict__ kh,
    const _Float16* __restrict__ vT, const float* __restrict__ gate,
    _Float16* __restrict__ yws) {
  __shared__ __align__(16) _Float16 plds[4][16 * 32];
  const int w    = threadIdx.x >> 5;
  const int lane = threadIdx.x & 31;
  const int h    = blockIdx.y * 4 + w;
  const int t0   = blockIdx.x * 16;
  const int col  = lane & 15;
  const int hi8  = (lane & 16) >> 1;

  const _Float16* qbase = qh + (h * T + t0) * HD;
  const v16h a0 = frag_a(qbase, HD);
  const v16h a1 = frag_a(qbase + 32, HD);

  v8f accy[4];
#pragma unroll
  for (int j = 0; j < 4; ++j) accy[j] = vzero8();
  float m_run[8], l_run[8];
#pragma unroll
  for (int r = 0; r < 8; ++r) { m_run[r] = -1e30f; l_run[r] = 0.f; }

  const int nblk = (t0 + 16 + 31) / 32;  // uniform across the block
  for (int sb = 0; sb < nblk; ++sb) {
    const int s0 = sb * 32;
    v8f sc[2];
#pragma unroll
    for (int ti = 0; ti < 2; ++ti) {
      const _Float16* kb = kh + (h * T + s0 + ti * 16) * HD;
      v8f z = vzero8();
      z = wmma16(a0, frag_a(kb, HD), z);
      z = wmma16(a1, frag_a(kb + 32, HD), z);
      sc[ti] = z;
    }
    // scale + causal mask + row max
    float mx[8];
#pragma unroll
    for (int r = 0; r < 8; ++r) {
      const int t = t0 + r + hi8;
      float e0 = sc[0][r] * 0.1f;
      float e1 = sc[1][r] * 0.1f;
      if (s0 + col > t)      e0 = -1e30f;
      if (s0 + 16 + col > t) e1 = -1e30f;
      sc[0][r] = e0; sc[1][r] = e1;
      mx[r] = fmaxf(e0, e1);
    }
#pragma unroll
    for (int msk = 1; msk < 16; msk <<= 1) {
#pragma unroll
      for (int r = 0; r < 8; ++r) mx[r] = fmaxf(mx[r], __shfl_xor(mx[r], msk, 32));
    }
    float rs[8], scl[8];
#pragma unroll
    for (int r = 0; r < 8; ++r) {
      const float m_new = fmaxf(m_run[r], mx[r]);
      scl[r]   = __expf(m_run[r] - m_new);
      m_run[r] = m_new;
      const float p0 = __expf(sc[0][r] - m_new);
      const float p1 = __expf(sc[1][r] - m_new);
      plds[w][(r + hi8) * 32 + col]      = (_Float16)p0;
      plds[w][(r + hi8) * 32 + 16 + col] = (_Float16)p1;
      rs[r] = p0 + p1;
#pragma unroll
      for (int j = 0; j < 4; ++j) accy[j][r] *= scl[r];
    }
#pragma unroll
    for (int msk = 1; msk < 16; msk <<= 1) {
#pragma unroll
      for (int r = 0; r < 8; ++r) rs[r] += __shfl_xor(rs[r], msk, 32);
    }
#pragma unroll
    for (int r = 0; r < 8; ++r) l_run[r] = l_run[r] * scl[r] + rs[r];

    __syncthreads();                       // P visible (cross-lane) before read
    const v16h pf = frag_a(&plds[w][0], 32);
#pragma unroll
    for (int j = 0; j < 4; ++j) {
      const v16h bv = frag_a(vT + (h * HD + j * 16) * T + s0, T);
      accy[j] = wmma16(pf, bv, accy[j]);
    }
    __syncthreads();                       // reads done before next overwrite
  }

  // y = (acc / l) * gate, stored f16 at [t][h*64+d] for the projection GEMM
#pragma unroll
  for (int r = 0; r < 8; ++r) {
    const int   t   = t0 + r + hi8;
    const float inv = 1.f / l_run[r];
    const float g   = gate[t * H + h];
#pragma unroll
    for (int j = 0; j < 4; ++j) {
      const float yv = accy[j][r] * inv * g;
      yws[t * D + h * HD + j * 16 + col] = (_Float16)yv;
    }
  }
}

// ---------------- Phase 3: output projection  out = y @ Wproj.T -------------
// y tile (16x1024 f16, 32 KB) shared by the 4 waves -> async-copied to LDS.
__global__ void __launch_bounds__(128) proj_kernel(
    const _Float16* __restrict__ yws, const _Float16* __restrict__ wp,
    float* __restrict__ out) {
  __shared__ __align__(16) _Float16 ytile[16 * D];   // 32 KB
  const int w    = threadIdx.x >> 5;
  const int lane = threadIdx.x & 31;
  const int t0   = blockIdx.x * 16;
  const int o0   = (blockIdx.y * 4 + w) * 64;
  const int col  = lane & 15;
  const int hi8  = (lane & 16) >> 1;

  async_stage(yws + t0 * D, &ytile[0], 16 * D * (int)sizeof(_Float16));
  __syncthreads();

  v8f acc[4];
#pragma unroll
  for (int j = 0; j < 4; ++j) acc[j] = vzero8();

  for (int k0 = 0; k0 < D; k0 += 32) {
    const v16h a = frag_a(ytile + k0, D);
#pragma unroll
    for (int j = 0; j < 4; ++j) {
      const v16h b = frag_a(wp + (o0 + j * 16) * D + k0, D);
      acc[j] = wmma16(a, b, acc[j]);
    }
  }
#pragma unroll
  for (int j = 0; j < 4; ++j) {
#pragma unroll
    for (int r = 0; r < 8; ++r) {
      out[(t0 + r + hi8) * D + o0 + j * 16 + col] = acc[j][r];
    }
  }
}

}  // namespace

extern "C" void kernel_launch(void* const* d_in, const int* in_sizes, int n_in,
                              void* d_out, int out_size, void* d_ws, size_t ws_size,
                              hipStream_t stream) {
  const float* x    = (const float*)d_in[0];
  const float* v1   = (const float*)d_in[1];
  const float* Wq   = (const float*)d_in[2];
  const float* Wk   = (const float*)d_in[3];
  const float* Wv   = (const float*)d_in[4];
  const float* Wp   = (const float*)d_in[5];
  const float* lamb = (const float*)d_in[6];
  const float* Wg   = (const float*)d_in[7];
  float* out = (float*)d_out;

  char* ws = (char*)d_ws;
  const size_t MB = 1024 * 1024;
  _Float16* xh   = (_Float16*)(ws + 0 * MB);    // 4 MB  x (f16)
  _Float16* wq16 = (_Float16*)(ws + 4 * MB);    // 2 MB
  _Float16* wk16 = (_Float16*)(ws + 6 * MB);    // 2 MB
  _Float16* wv16 = (_Float16*)(ws + 8 * MB);    // 2 MB
  _Float16* wp16 = (_Float16*)(ws + 10 * MB);   // 2 MB
  _Float16* qh   = (_Float16*)(ws + 12 * MB);   // 4 MB  [h][t][64]
  _Float16* kh   = (_Float16*)(ws + 16 * MB);   // 4 MB  [h][s][64]
  _Float16* vT   = (_Float16*)(ws + 20 * MB);   // 4 MB  [h][d][s]
  _Float16* yws  = (_Float16*)(ws + 24 * MB);   // 4 MB  [t][dim]
  float*    gate = (float*)   (ws + 28 * MB);   // 128 KB [t][h]

  const int n_x = T * D, n_w = D * D;
  convert_f16<<<(n_x + 255) / 256, 256, 0, stream>>>(x,  xh,   n_x);
  convert_f16<<<(n_w + 255) / 256, 256, 0, stream>>>(Wq, wq16, n_w);
  convert_f16<<<(n_w + 255) / 256, 256, 0, stream>>>(Wk, wk16, n_w);
  convert_f16<<<(n_w + 255) / 256, 256, 0, stream>>>(Wv, wv16, n_w);
  convert_f16<<<(n_w + 255) / 256, 256, 0, stream>>>(Wp, wp16, n_w);

  qkv_kernel <<<dim3(T / 16, 4), 128, 0, stream>>>(x, v1, lamb, Wg, xh, wq16,
                                                   wk16, wv16, qh, kh, vT, gate);
  attn_kernel<<<dim3(T / 16, 4), 128, 0, stream>>>(qh, kh, vT, gate, yws);
  proj_kernel<<<dim3(T / 16, 4), 128, 0, stream>>>(yws, wp16, out);

  // second tuple output: v1 passthrough
  hipMemcpyAsync(out + (size_t)T * D, v1, (size_t)T * D * sizeof(float),
                 hipMemcpyDeviceToDevice, stream);
}
